// ShowAttendAndTell_53558242181145
// MI455X (gfx1250) — compile-verified, matrix operand order checked by reading, not run on hardware
//
#include <hip/hip_runtime.h>
#include <hip/hip_bf16.h>

// ---------------------------------------------------------------------------
// Show-Attend-and-Tell decoder for MI455X (gfx1250, wave32, WMMA).
// GEMMs run on v_wmma_f32_16x16x32_f16 with 2x2 register blocking per wave
// (32x32 output tile -> 4 WMMAs per K-step, 1.0 fragment-loads per WMMA).
// feat_proj is materialized once in fp16 (67MB -> L2-resident across steps).
// ---------------------------------------------------------------------------

#define B_ 256
#define P_ 256
#define T_ 32
#define V_ 3433
#define VP_ 3456   // V padded to multiple of 32 (2x2 tile blocking)
#define H_ 512
#define E_ 300
#define EP_ 320    // E padded to multiple of 32 (K granularity)
#define C_ 128
#define NS_ 31     // n_steps = T-1 (static, matches reference trace)

typedef __attribute__((ext_vector_type(16))) _Float16 v16h;
typedef __attribute__((ext_vector_type(8)))  _Float16 v8h;
typedef __attribute__((ext_vector_type(8)))  float    v8f;

// A-fragment (16-bit A 16x32, ISA 7.12.2): lane L holds row L%16,
// K = {kb..kb+7} U {kb+16..kb+23} with kb = 8*(L/16). W loads identically.
__device__ __forceinline__ v16h load_frag(const _Float16* p) {
  v8h lo = *reinterpret_cast<const v8h*>(p);
  v8h hi = *reinterpret_cast<const v8h*>(p + 16);
  v16h r;
#pragma unroll
  for (int i = 0; i < 8; ++i) { r[i] = lo[i]; r[i + 8] = hi[i]; }
  return r;
}

__device__ __forceinline__ v8f wmma_f16(v16h a, v16h b, v8f c) {
  return __builtin_amdgcn_wmma_f32_16x16x32_f16(
      /*neg_a=*/false, a, /*neg_b=*/false, b,
      /*c_mod=*/(short)0, c, /*reuse_a=*/false, /*reuse_b=*/false);
}

// ---------------------------------------------------------------------------
// Generic WMMA GEMM:  C[M,N] = act( A[M,K] * W[N,K]^T + bias[N] + addend[M,N] )
// act: 0=none 1=tanh 2=relu.  Optional fp32 and/or fp16 outputs.
// Each wave computes one 32x32 tile (2x2 of 16x16); 8 waves / block.
// ---------------------------------------------------------------------------
__global__ __launch_bounds__(256) void wmma_gemm(
    const _Float16* __restrict__ A,  int lda,
    const _Float16* __restrict__ Bw, int ldb,
    const float* __restrict__ bias,
    const float* __restrict__ addend, int ldadd,
    float* __restrict__ Cf, int ldc,
    _Float16* __restrict__ Ch, int ldch,
    int M, int N, int Nreal, int K, int act)
{
  const int tileM = M >> 5;
  const int tileN = N >> 5;
  const int wave  = threadIdx.x >> 5;
  const int lane  = threadIdx.x & 31;
  const int tile  = blockIdx.x * 8 + wave;
  if (tile >= tileM * tileN) return;          // wave-uniform: EXEC stays all-1
  const int tm   = (tile % tileM) << 5;
  const int tn   = (tile / tileM) << 5;
  const int row  = lane & 15;
  const int koff = (lane >> 4) << 3;          // 0 or 8

  const _Float16* a0 = A  + (size_t)(tm + row) * lda + koff;
  const _Float16* a1 = a0 + (size_t)16 * lda;
  const _Float16* b0 = Bw + (size_t)(tn + row) * ldb + koff;
  const _Float16* b1 = b0 + (size_t)16 * ldb;

  v8f acc00 = {}, acc01 = {}, acc10 = {}, acc11 = {};
  for (int k = 0; k < K; k += 32) {
    __builtin_prefetch(b0 + k + 256, 0, 1);   // global_prefetch_b8 (weights)
    __builtin_prefetch(b1 + k + 256, 0, 1);
    v16h fa0 = load_frag(a0 + k);
    v16h fa1 = load_frag(a1 + k);
    v16h fb0 = load_frag(b0 + k);
    v16h fb1 = load_frag(b1 + k);
    acc00 = wmma_f16(fa0, fb0, acc00);
    acc01 = wmma_f16(fa0, fb1, acc01);
    acc10 = wmma_f16(fa1, fb0, acc10);
    acc11 = wmma_f16(fa1, fb1, acc11);
  }

  // D layout: lane L -> col L%16 ; vgpr r -> row 8*(L/16) + r  (per 16x16 tile)
  const int nlo = lane & 15;
  const int mb  = (lane >> 4) << 3;
  auto emit = [&](const v8f& acc, int mo, int no) {
    const int cn = tn + no + nlo;
    if (cn >= Nreal) return;
#pragma unroll
    for (int r = 0; r < 8; ++r) {
      const int m = tm + mo + mb + r;
      float v = acc[r];
      if (bias)   v += bias[cn];
      if (addend) v += addend[(size_t)m * ldadd + cn];
      if (act == 1)      v = tanhf(v);
      else if (act == 2) v = fmaxf(v, 0.0f);
      if (Cf) Cf[(size_t)m * ldc  + cn] = v;
      if (Ch) Ch[(size_t)m * ldch + cn] = (_Float16)v;
    }
  };
  emit(acc00, 0, 0);  emit(acc01, 0, 16);
  emit(acc10, 16, 0); emit(acc11, 16, 16);
}

// fp32 -> fp16 with optional row/col zero padding
__global__ void f32_to_f16_pad(const float* __restrict__ src, _Float16* __restrict__ dst,
                               int rowsSrc, int rowsDst, int ks, int kd)
{
  size_t idx   = (size_t)blockIdx.x * blockDim.x + threadIdx.x;
  size_t total = (size_t)rowsDst * kd;
  if (idx >= total) return;
  int r = (int)(idx / kd), c = (int)(idx % kd);
  float v = (r < rowsSrc && c < ks) ? src[(size_t)r * ks + c] : 0.0f;
  dst[idx] = (_Float16)v;
}

__global__ void zero_f32(float* p, int n) {
  int i = blockIdx.x * blockDim.x + threadIdx.x;
  if (i < n) p[i] = 0.0f;
}
__global__ void zero_f16(_Float16* p, int n) {
  int i = blockIdx.x * blockDim.x + threadIdx.x;
  if (i < n) p[i] = (_Float16)0.0f;
}

// Fused GRU gates: h = (1-z)*n + z*h ; also emits fp16 mirror for next GEMMs
__global__ void gru_update(const float* __restrict__ gx, const float* __restrict__ gh,
                           float* __restrict__ h, _Float16* __restrict__ hf)
{
  int idx = blockIdx.x * blockDim.x + threadIdx.x;   // B*H threads
  int b = idx >> 9, j = idx & 511;
  size_t g = (size_t)b * (3 * H_);
  float xr = gx[g + j], xz = gx[g + H_ + j], xn = gx[g + 2 * H_ + j];
  float hr = gh[g + j], hz = gh[g + H_ + j], hn = gh[g + 2 * H_ + j];
  float hv = h[idx];
  float r = 1.0f / (1.0f + expf(-(xr + hr)));
  float z = 1.0f / (1.0f + expf(-(xz + hz)));
  float n = tanhf(xn + r * hn);
  float out = (1.0f - z) * n + z * hv;
  h[idx]  = out;
  hf[idx] = (_Float16)out;
}

// scores[b,p] = sum_h W_att[h] * tanh(feat_proj16[b,p,h] + hW[b,h])
// block = one batch row; wave = one proposal at a time; shfl reduction.
__global__ __launch_bounds__(256) void attn_scores(
    const _Float16* __restrict__ fproj, const float* __restrict__ hw,
    const float* __restrict__ watt, float* __restrict__ scores)
{
  __shared__ float s_hw[H_];
  __shared__ float s_wa[H_];
  int b = blockIdx.x, tid = threadIdx.x;
  s_hw[tid]       = hw[(size_t)b * H_ + tid];
  s_hw[tid + 256] = hw[(size_t)b * H_ + 256 + tid];
  s_wa[tid]       = watt[tid];
  s_wa[tid + 256] = watt[256 + tid];
  __syncthreads();
  int wave = tid >> 5, lane = tid & 31;
  for (int p = wave; p < P_; p += 8) {
    const _Float16* fp = fproj + (((size_t)b * P_ + p) << 9) + lane * 16;
    v8h u0 = *reinterpret_cast<const v8h*>(fp);
    v8h u1 = *reinterpret_cast<const v8h*>(fp + 8);
    float acc = 0.0f;
#pragma unroll
    for (int i = 0; i < 8; ++i) {
      int hh = lane * 16 + i;
      acc += s_wa[hh]     * tanhf((float)u0[i] + s_hw[hh]);
      acc += s_wa[hh + 8] * tanhf((float)u1[i] + s_hw[hh + 8]);
    }
#pragma unroll
    for (int off = 16; off; off >>= 1) acc += __shfl_xor(acc, off, 32);
    if (lane == 0) scores[b * P_ + p] = acc;
  }
}

// softmax over P, emit mask into d_out attn section, then attended = mask @ c_feats
__global__ __launch_bounds__(256) void softmax_attend(
    const float* __restrict__ scores, const float* __restrict__ cfeats,
    float* __restrict__ attended, _Float16* __restrict__ att16,
    float* __restrict__ attn_out, int t)
{
  __shared__ float sm[P_];
  __shared__ float red[P_];
  int b = blockIdx.x, p = threadIdx.x;
  float v = scores[b * P_ + p];
  red[p] = v; __syncthreads();
  for (int s = 128; s; s >>= 1) { if (p < s) red[p] = fmaxf(red[p], red[p + s]); __syncthreads(); }
  float mx = red[0]; __syncthreads();
  float e = expf(v - mx);
  red[p] = e; __syncthreads();
  for (int s = 128; s; s >>= 1) { if (p < s) red[p] += red[p + s]; __syncthreads(); }
  float mask = e / red[0];
  sm[p] = mask;
  attn_out[((size_t)b * P_ + p) * NS_ + t] = mask;  // (B,P,T-1)
  __syncthreads();
  if (p < C_) {
    const float* cf = cfeats + (size_t)b * P_ * C_ + p;
    float a = 0.0f;
#pragma unroll 4
    for (int pp = 0; pp < P_; ++pp) a += sm[pp] * cf[(size_t)pp * C_];
    attended[b * C_ + p] = a;
    att16[b * C_ + p]    = (_Float16)a;
  }
}

// ---------------------------------------------------------------------------
static inline void gemm(hipStream_t s,
                        const _Float16* A, int lda, const _Float16* Bw, int ldb,
                        const float* bias, const float* addend, int ldadd,
                        float* Cf, int ldc, _Float16* Ch, int ldch,
                        int M, int N, int Nreal, int K, int act)
{
  int tiles = (M / 32) * (N / 32);
  int blocks = (tiles + 7) / 8;
  wmma_gemm<<<blocks, 256, 0, s>>>(A, lda, Bw, ldb, bias, addend, ldadd,
                                   Cf, ldc, Ch, ldch, M, N, Nreal, K, act);
}

static inline void conv(hipStream_t s, const float* src, _Float16* dst,
                        int rowsSrc, int rowsDst, int ks, int kd)
{
  size_t total = (size_t)rowsDst * kd;
  int blocks = (int)((total + 255) / 256);
  f32_to_f16_pad<<<blocks, 256, 0, s>>>(src, dst, rowsSrc, rowsDst, ks, kd);
}

extern "C" void kernel_launch(void* const* d_in, const int* in_sizes, int n_in,
                              void* d_out, int out_size, void* d_ws, size_t ws_size,
                              hipStream_t stream)
{
  (void)in_sizes; (void)n_in; (void)out_size;
  const float* word_embs = (const float*)d_in[0];
  const float* t_feat    = (const float*)d_in[1];
  const float* c_feats   = (const float*)d_in[2];
  const float* W_td1 = (const float*)d_in[4];
  const float* W_td2 = (const float*)d_in[5];
  const float* W_td3 = (const float*)d_in[6];
  const float* W_td  = (const float*)d_in[7];
  const float* W1_ih = (const float*)d_in[8];
  const float* W1_hh = (const float*)d_in[9];
  const float* b1_ih = (const float*)d_in[10];
  const float* b1_hh = (const float*)d_in[11];
  const float* W_feat = (const float*)d_in[12];
  const float* W_hidd = (const float*)d_in[13];
  const float* W_att  = (const float*)d_in[14];
  const float* W_l1 = (const float*)d_in[15];
  const float* W_l2 = (const float*)d_in[16];
  const float* W_l  = (const float*)d_in[17];
  const float* W2_ih = (const float*)d_in[18];
  const float* W2_hh = (const float*)d_in[19];
  const float* b2_ih = (const float*)d_in[20];
  const float* b2_hh = (const float*)d_in[21];
  const float* W_cls = (const float*)d_in[22];
  const float* b_cls = (const float*)d_in[23];

  float* out      = (float*)d_out;
  float* attn_out = out + (size_t)B_ * NS_ * V_;

  // ---- workspace carve-out (256B aligned) --------------------------------
  uint8_t* w = (uint8_t*)d_ws;
  size_t off = 0;
  auto A16 = [&](size_t n) { size_t r = off; off = (off + n * 2 + 255) & ~(size_t)255; return (_Float16*)(w + r); };
  auto A32 = [&](size_t n) { size_t r = off; off = (off + n * 4 + 255) & ~(size_t)255; return (float*)(w + r); };

  _Float16* we16   = A16((size_t)B_ * T_ * EP_);
  _Float16* tf16   = A16((size_t)B_ * C_);
  _Float16* c16    = A16((size_t)B_ * P_ * C_);
  _Float16* fproj  = A16((size_t)B_ * P_ * H_);       // 67MB, L2-resident
  _Float16* Wtd1f  = A16(128 * 512);
  _Float16* Wtd2f  = A16(128 * 128);
  _Float16* Wtd3f  = A16(128 * EP_);
  _Float16* Wtdf   = A16(128 * 128);
  _Float16* W1ihf  = A16(1536 * 128);
  _Float16* W1hhf  = A16((size_t)1536 * 512);
  _Float16* Wfeatf = A16(512 * 128);
  _Float16* Whiddf = A16((size_t)512 * 512);
  _Float16* Wl1f   = A16(128 * 128);
  _Float16* Wl2f   = A16(128 * 512);
  _Float16* Wlf    = A16(128 * 128);
  _Float16* W2ihf  = A16(1536 * 128);
  _Float16* W2hhf  = A16((size_t)1536 * 512);
  _Float16* Wclsf  = A16((size_t)VP_ * 512);
  float* td_t  = A32((size_t)B_ * 128);
  float* h1    = A32((size_t)B_ * H_);
  float* h2    = A32((size_t)B_ * H_);
  _Float16* h1f = A16((size_t)B_ * H_);
  _Float16* h2f = A16((size_t)B_ * H_);
  float* gx    = A32((size_t)B_ * 3 * H_);
  float* gh    = A32((size_t)B_ * 3 * H_);
  float* tmpA  = A32((size_t)B_ * 128);
  _Float16* st16 = A16((size_t)B_ * 128);
  _Float16* s16  = A16((size_t)B_ * 128);
  _Float16* lt16 = A16((size_t)B_ * 128);
  _Float16* l16  = A16((size_t)B_ * 128);
  _Float16* at16 = A16((size_t)B_ * 128);
  float* hw      = A32((size_t)B_ * H_);
  float* scores  = A32((size_t)B_ * P_);
  float* attnd   = A32((size_t)B_ * 128);
  if (off > ws_size) return;   // insufficient scratch: deterministic no-op

  // ---- one-time conversions / precompute ---------------------------------
  conv(stream, word_embs, we16, B_ * T_, B_ * T_, E_, EP_);
  conv(stream, t_feat,    tf16, B_, B_, C_, C_);
  conv(stream, c_feats,   c16,  B_ * P_, B_ * P_, C_, C_);
  conv(stream, W_td1, Wtd1f, 128, 128, 512, 512);
  conv(stream, W_td2, Wtd2f, 128, 128, 128, 128);
  conv(stream, W_td3, Wtd3f, 128, 128, E_, EP_);
  conv(stream, W_td,  Wtdf,  128, 128, 128, 128);
  conv(stream, W1_ih, W1ihf, 1536, 1536, 128, 128);
  conv(stream, W1_hh, W1hhf, 1536, 1536, 512, 512);
  conv(stream, W_feat, Wfeatf, 512, 512, 128, 128);
  conv(stream, W_hidd, Whiddf, 512, 512, 512, 512);
  conv(stream, W_l1, Wl1f, 128, 128, 128, 128);
  conv(stream, W_l2, Wl2f, 128, 128, 512, 512);
  conv(stream, W_l,  Wlf,  128, 128, 128, 128);
  conv(stream, W2_ih, W2ihf, 1536, 1536, 128, 128);
  conv(stream, W2_hh, W2hhf, 1536, 1536, 512, 512);
  conv(stream, W_cls, Wclsf, V_, VP_, 512, 512);     // row-padded to 3456

  zero_f32<<<(B_ * H_) / 256, 256, 0, stream>>>(h1, B_ * H_);
  zero_f32<<<(B_ * H_) / 256, 256, 0, stream>>>(h2, B_ * H_);
  zero_f16<<<(B_ * H_) / 256, 256, 0, stream>>>(h1f, B_ * H_);
  zero_f16<<<(B_ * H_) / 256, 256, 0, stream>>>(h2f, B_ * H_);

  // td_t = t_feat @ W_td2^T        (B,128)
  gemm(stream, tf16, C_, Wtd2f, C_, nullptr, nullptr, 0,
       td_t, 128, nullptr, 0, B_, 128, 128, C_, 0);
  // feat_proj = c_feats @ W_feat^T (B*P,H) in fp16
  gemm(stream, c16, C_, Wfeatf, C_, nullptr, nullptr, 0,
       nullptr, 0, fproj, H_, B_ * P_, H_, H_, C_, 0);

  // ---- 31 decode steps ---------------------------------------------------
  for (int t = 0; t < NS_; ++t) {
    // top-down unit
    gemm(stream, we16 + (size_t)t * EP_, T_ * EP_, Wtd3f, EP_, nullptr,
         td_t, 128, tmpA, 128, nullptr, 0, B_, 128, 128, EP_, 0);
    gemm(stream, h2f, H_, Wtd1f, H_, nullptr, tmpA, 128,
         nullptr, 0, st16, 128, B_, 128, 128, H_, 1);            // tanh
    gemm(stream, st16, 128, Wtdf, 128, nullptr, nullptr, 0,
         nullptr, 0, s16, 128, B_, 128, 128, 128, 2);            // relu
    // GRU 1
    gemm(stream, s16, 128, W1ihf, 128, b1_ih, nullptr, 0,
         gx, 3 * H_, nullptr, 0, B_, 3 * H_, 3 * H_, 128, 0);
    gemm(stream, h1f, H_, W1hhf, H_, b1_hh, nullptr, 0,
         gh, 3 * H_, nullptr, 0, B_, 3 * H_, 3 * H_, H_, 0);
    gru_update<<<(B_ * H_) / 256, 256, 0, stream>>>(gx, gh, h1, h1f);
    // additive attention
    gemm(stream, h1f, H_, Whiddf, H_, nullptr, nullptr, 0,
         hw, H_, nullptr, 0, B_, H_, H_, H_, 0);
    attn_scores<<<B_, 256, 0, stream>>>(fproj, hw, W_att, scores);
    softmax_attend<<<B_, 256, 0, stream>>>(scores, c_feats, attnd, at16, attn_out, t);
    // language unit
    gemm(stream, at16, C_, Wl1f, C_, nullptr, nullptr, 0,
         tmpA, 128, nullptr, 0, B_, 128, 128, C_, 0);
    gemm(stream, h1f, H_, Wl2f, H_, nullptr, tmpA, 128,
         nullptr, 0, lt16, 128, B_, 128, 128, H_, 1);            // tanh
    gemm(stream, lt16, 128, Wlf, 128, nullptr, nullptr, 0,
         nullptr, 0, l16, 128, B_, 128, 128, 128, 2);            // relu
    // GRU 2
    gemm(stream, l16, 128, W2ihf, 128, b2_ih, nullptr, 0,
         gx, 3 * H_, nullptr, 0, B_, 3 * H_, 3 * H_, 128, 0);
    gemm(stream, h2f, H_, W2hhf, H_, b2_hh, nullptr, 0,
         gh, 3 * H_, nullptr, 0, B_, 3 * H_, 3 * H_, H_, 0);
    gru_update<<<(B_ * H_) / 256, 256, 0, stream>>>(gx, gh, h2, h2f);
    // classifier -> lang_cap[:, t, :]  (ldc strides over the (T-1,V) plane)
    gemm(stream, h2f, H_, Wclsf, H_, b_cls, nullptr, 0,
         out + (size_t)t * V_, NS_ * V_, nullptr, 0,
         B_, VP_, V_, H_, 0);
  }
}
